// MGMP_23373212025295
// MI455X (gfx1250) — matrix operand order repaired and checked by reading, not compiled, not run on hardware
//
#include <hip/hip_runtime.h>
#include <hip/hip_bf16.h>

#define B_ 32
#define L_ 256
#define K_ 50
#define S_ 16
#define D_ 128
#define H_ 4
#define C_ 128
#define NN_ 100000

typedef __attribute__((ext_vector_type(16))) _Float16 v16h;
typedef __attribute__((ext_vector_type(8)))  float    v8f;
typedef __attribute__((ext_vector_type(4)))  unsigned int ui32x4;
typedef __attribute__((ext_vector_type(8)))  int      i32x8;
typedef __attribute__((ext_vector_type(4)))  int      i32x4;

// Tensor Data Mover availability (compile-guarded; falls back to plain gathers).
#if defined(__has_builtin)
#if __has_builtin(__builtin_amdgcn_tensor_load_to_lds) && __has_builtin(__builtin_amdgcn_s_wait_tensorcnt)
#define CTX_TDM 1
#endif
#endif
#ifndef CTX_TDM
#define CTX_TDM 0
#endif

// ---------------- WMMA fragment helpers (CDNA5 16x16x32 f16 layouts) -----------

// A 16x32 f16 tile from LDS (row-major, leading dim lda): two contiguous 8-half runs.
__device__ __forceinline__ v16h load_a_lds(const _Float16* A, int lda, int mbase,
                                           int kbase, int lane) {
  const int m = lane & 15, hl = lane >> 4;
  const _Float16* p0 = A + (long)(mbase + m) * lda + kbase + hl * 8;
  const _Float16* p1 = p0 + 16;
  v16h a;
#pragma unroll
  for (int i = 0; i < 8; ++i) { a[i] = p0[i]; a[8 + i] = p1[i]; }
  return a;
}

// A tile from global f16 (im2col over doc rows), clamped runs.
__device__ __forceinline__ v16h load_a_doc(const _Float16* A, int mbase, int kbase,
                                           long maxs, int lane) {
  const int m = lane & 15, hl = lane >> 4;
  long i0 = (long)(mbase + m) * D_ + kbase + hl * 8;
  long i1 = i0 + 16;
  if (i0 > maxs) i0 = maxs;
  if (i1 > maxs) i1 = maxs;
  v16h a;
#pragma unroll
  for (int i = 0; i < 8; ++i) { a[i] = A[i0 + i]; a[8 + i] = A[i1 + i]; }
  return a;
}

// B 32x16 f16 tile for Y = X @ W^T from pre-converted f16 weights:
// B[k][n] = W[nbase+n][kbase+k]; per-lane two contiguous 8-half runs (2x b128).
__device__ __forceinline__ v16h load_b_f16(const _Float16* W, int ldk, int nbase,
                                           int kbase, int lane) {
  const int n = lane & 15, hl = lane >> 4;
  const _Float16* p = W + (long)(nbase + n) * ldk + kbase + hl * 8;
  v16h b;
#pragma unroll
  for (int i = 0; i < 8; ++i) { b[i] = p[i]; b[8 + i] = p[16 + i]; }
  return b;
}

#define WMMA_F16(acc, a, b) \
  __builtin_amdgcn_wmma_f32_16x16x32_f16(false, (a), false, (b), (short)0, (acc), false, false)

// ---------------- weight pre-conversion kernels --------------------------------
__global__ __launch_bounds__(256) void cvt_f16_kernel(const float* __restrict__ s,
                                                      _Float16* __restrict__ d, int n) {
  const int i = blockIdx.x * 256 + threadIdx.x;
  if (i < n) d[i] = (_Float16)s[i];
}

// conv w (C, D, kt) -> Wp[c][k] with k = t*D + d  (im2col B layout, contiguous k)
__global__ __launch_bounds__(256) void repack_conv_kernel(const float* __restrict__ w,
                                                          _Float16* __restrict__ d, int kt) {
  const int i = blockIdx.x * 256 + threadIdx.x;
  const int tot = C_ * D_ * kt;
  if (i < tot) {
    const int c = i / (D_ * kt), r = i % (D_ * kt);
    const int dd = r / kt, t = r % kt;
    d[c * (D_ * kt) + t * D_ + dd] = (_Float16)w[i];
  }
}

// ---------------- Kernel 1: word-level fine attention -> doc (f16) -------------
__global__ __launch_bounds__(256) void fine_attn_kernel(
    const int* __restrict__ nwi, const int* __restrict__ wctx,
    const float* __restrict__ word_table, const float* __restrict__ news_table,
    const float* __restrict__ in_w, const float* __restrict__ in_b,
    const _Float16* __restrict__ kv16,  // [2][128][128] f16 (K then V rows of in_w)
    const float* __restrict__ out_w, const float* __restrict__ out_b,
    _Float16* __restrict__ doch) {
  __shared__ _Float16 ctxh[64][D_];
  __shared__ union KVU {              // f32 TDM staging aliases kh/vh f16 buffers
    float    stage[64][D_];
    _Float16 kv[2][64][D_];
  } u;
  __shared__ float qh[D_], ov[D_];
  __shared__ float sc[H_][K_];
#if !CTX_TDM
  __shared__ int cid[64];
#endif

  const int tok = blockIdx.x;  // b*L + l
  const int tid = threadIdx.x, lane = tid & 31, wave = tid >> 5;
  const int widx = nwi[tok];

  if (tid == 0) __builtin_prefetch(word_table + (long)widx * D_, 0, 0);

#if CTX_TDM
  // TDM gather: 50 ctx rows of news_table -> LDS stage, 8 rows per descriptor.
  if (wave == 0) {
    const unsigned lds_base = (unsigned)(size_t)(&u.stage[0][0]);
    const unsigned long long gaddr = (unsigned long long)(size_t)news_table;
    for (int g = 0; g < 7; ++g) {
      const int r0 = g * 8;
      const int nidx = (K_ - r0) < 8 ? (K_ - r0) : 8;
      int id[8];
#pragma unroll
      for (int j = 0; j < 8; ++j) id[j] = (j < nidx) ? wctx[tok * K_ + r0 + j] : 0;
      ui32x4 g0;
      g0[0] = 1u | (1u << 30) | (1u << 31);            // count=1, 32b idx, gather
      g0[1] = lds_base + (unsigned)(r0 * D_ * 4);      // lds_addr
      g0[2] = (unsigned)(gaddr & 0xFFFFFFFFull);       // global_addr lo
      g0[3] = (unsigned)((gaddr >> 32) & 0x01FFFFFFull) | (2u << 30);  // hi | type=2
      i32x8 g1;
      g1[0] = (int)(2u << 16);                          // data_size = 4B
      g1[1] = (int)(128u << 16);                        // tensor_dim0 = 128
      g1[2] = (int)((unsigned)(NN_ & 0xFFFF) << 16);    // tensor_dim1 lo16
      g1[3] = (int)((128u << 16) | ((NN_ >> 16) & 0xFFFF));  // tile_dim0 | dim1 hi
      g1[4] = nidx;                                     // tile_dim1 = #indices
      g1[5] = 128;                                      // tensor_dim0_stride
      g1[6] = 0; g1[7] = 0;
      i32x4 g2, g3;
      g2[0] = id[0]; g2[1] = id[1]; g2[2] = id[2]; g2[3] = id[3];
      g3[0] = id[4]; g3[1] = id[5]; g3[2] = id[6]; g3[3] = id[7];
      i32x8 gpad = {};
      __builtin_amdgcn_tensor_load_to_lds(g0, g1, g2, g3, gpad, 0);
    }
  }
#else
  if (tid < K_) cid[tid] = wctx[tok * K_ + tid];
#endif

  // q projection matvec overlaps the DMA (reads straight from global, no LDS dep)
  if (tid < D_) {
    const float* wr = in_w + (long)tid * D_;
    const float* qrow = word_table + (long)widx * D_;
    float acc = in_b[tid];
    for (int d = 0; d < D_; ++d) acc += qrow[d] * wr[d];
    qh[tid] = acc;
  }

#if CTX_TDM
  if (wave == 0) __builtin_amdgcn_s_wait_tensorcnt(0);
  __syncthreads();
  for (int idx = tid; idx < 64 * D_; idx += 256) {
    const int r = idx >> 7, c = idx & 127;
    ctxh[r][c] = (r < K_) ? (_Float16)u.stage[r][c] : (_Float16)0.f;
  }
#else
  __syncthreads();
  for (int idx = tid; idx < 64 * D_; idx += 256) {
    const int r = idx >> 7, c = idx & 127;
    const float v = (r < K_) ? news_table[(long)cid[r] * D_ + c] : 0.f;
    ctxh[r][c] = (_Float16)v;
  }
#endif
  __syncthreads();

  // K and V projections: ctx(64x128) @ W^T via WMMA, wave -> 16-col N tile
  const int nb = wave * 16;
  const int n = lane & 15;
#pragma unroll
  for (int proj = 0; proj < 2; ++proj) {
    const _Float16* W = kv16 + (long)proj * D_ * D_;
    const float bias = in_b[D_ + proj * D_ + nb + n];
    for (int mt = 0; mt < 4; ++mt) {
      v8f acc = {};
#pragma unroll
      for (int kk = 0; kk < 4; ++kk) {
        v16h a = load_a_lds(&ctxh[0][0], D_, mt * 16, kk * 32, lane);
        v16h bf = load_b_f16(W, D_, nb, kk * 32, lane);
        acc = WMMA_F16(acc, a, bf);
      }
#pragma unroll
      for (int v = 0; v < 8; ++v) {
        const int m = mt * 16 + v + 8 * (lane >> 4);
        u.kv[proj][m][nb + n] = (_Float16)(acc[v] + bias);
      }
    }
  }
  __syncthreads();

  // per-head scores over 50 keys
  for (int s = tid; s < H_ * K_; s += 256) {
    const int h = s / K_, j = s % K_;
    float a = 0.f;
    for (int d = 0; d < 32; ++d) a += qh[h * 32 + d] * (float)u.kv[0][j][h * 32 + d];
    sc[h][j] = a * 0.17677669529663687f;  // 1/sqrt(32)
  }
  __syncthreads();
  if (tid < H_) {
    float mx = -1e30f;
    for (int j = 0; j < K_; ++j) mx = fmaxf(mx, sc[tid][j]);
    float sum = 0.f;
    for (int j = 0; j < K_; ++j) { const float e = expf(sc[tid][j] - mx); sc[tid][j] = e; sum += e; }
    const float inv = 1.f / sum;
    for (int j = 0; j < K_; ++j) sc[tid][j] *= inv;
  }
  __syncthreads();
  if (tid < D_) {
    const int h = tid >> 5;
    float a = 0.f;
    for (int j = 0; j < K_; ++j) a += sc[h][j] * (float)u.kv[1][j][tid];
    ov[tid] = a;
  }
  __syncthreads();
  if (tid < D_) {
    const float* wr = out_w + (long)tid * D_;
    float acc = out_b[tid];
    for (int d = 0; d < D_; ++d) acc += ov[d] * wr[d];
    doch[(long)tok * D_ + tid] = (widx == 0) ? (_Float16)0.f : (_Float16)acc;
  }
}

// ---------------- Kernel 2: conv1d(3/4/5) + relu + maxpool + fc ----------------
__global__ __launch_bounds__(256) void conv_pool_fc_kernel(
    const _Float16* __restrict__ doch,
    const _Float16* __restrict__ cp3, const float* __restrict__ b3,
    const _Float16* __restrict__ cp4, const float* __restrict__ b4,
    const _Float16* __restrict__ cp5, const float* __restrict__ b5,
    const float* __restrict__ fc_w, const float* __restrict__ fc_b,
    float* __restrict__ semantic) {
  __shared__ float pooled[3 * C_];
  const int b = blockIdx.x, tid = threadIdx.x, lane = tid & 31, wave = tid >> 5;
  const _Float16* dflat = doch + (long)b * L_ * D_;
  const int nb = wave * 16;
  const long maxs = (long)L_ * D_ - 8;
  const _Float16* Ws[3] = {cp3, cp4, cp5};
  const float* Bs[3] = {b3, b4, b5};

  for (int ci = 0; ci < 3; ++ci) {
    const int kt = 3 + ci;
    const int Ktot = kt * D_;
    const _Float16* W = Ws[ci];
    const float bias = Bs[ci][nb + (lane & 15)];
    const int valid = L_ - kt + 1;
    float colmax = 0.f;  // relu >= 0, so 0 is a safe max identity
    for (int mt = 0; mt < 16; ++mt) {
      v8f acc = {};
      for (int kk = 0; kk < Ktot; kk += 32) {
        v16h a = load_a_doc(dflat, mt * 16, kk, maxs, lane);
        v16h bf = load_b_f16(W, Ktot, nb, kk, lane);
        acc = WMMA_F16(acc, a, bf);
      }
#pragma unroll
      for (int v = 0; v < 8; ++v) {
        const int p = mt * 16 + v + 8 * (lane >> 4);
        const float r = fmaxf(acc[v] + bias, 0.f);
        if (p < valid) colmax = fmaxf(colmax, r);
      }
    }
    colmax = fmaxf(colmax, __shfl_xor(colmax, 16));
    if (lane < 16) pooled[ci * C_ + nb + lane] = colmax;
  }
  __syncthreads();
  if (tid < D_) {
    const float* wr = fc_w + (long)tid * (3 * C_);
    float acc = fc_b[tid];
    for (int i = 0; i < 3 * C_; ++i) acc += pooled[i] * wr[i];
    semantic[b * D_ + tid] = acc;
  }
}

// ---------------- Kernel 3: meta-path GRU + MHA + LN + mean --------------------
// w16 block layout (halves): [wih 3D*D][whh 3D*D][in_w 3D*D][out_w D*D]
__global__ __launch_bounds__(256) void metapath_kernel(
    const int* __restrict__ news_ids, const int* __restrict__ mid_idx,
    const int* __restrict__ end_idx,
    const float* __restrict__ news_table, const float* __restrict__ mid_table,
    const _Float16* __restrict__ w16,
    const float* __restrict__ gru_bih, const float* __restrict__ gru_bhh,
    const float* __restrict__ in_b, const float* __restrict__ out_b,
    const float* __restrict__ ln_g, const float* __restrict__ ln_b,
    float* __restrict__ eout) {
  __shared__ _Float16 x16[S_][D_];
  __shared__ float hbuf[S_][D_];
  __shared__ _Float16 h16[S_][D_];
  __shared__ float G[S_][3 * D_];
  __shared__ float hn[S_][D_];
  __shared__ float sc[H_ * S_ * S_];
  __shared__ _Float16 o16[S_][D_];
  __shared__ float mu[S_], rs[S_];

  const int b = blockIdx.x, tid = threadIdx.x, lane = tid & 31, wave = tid >> 5;
  const int n = lane & 15;
  const _Float16* wih16 = w16;
  const _Float16* whh16 = w16 + 3 * D_ * D_;
  const _Float16* inw16 = w16 + 6 * D_ * D_;
  const _Float16* outw16 = w16 + 9 * D_ * D_;

  for (int i = tid; i < S_ * D_; i += 256) (&hbuf[0][0])[i] = 0.f;
  __syncthreads();

  // ---- 3-step GRU over the path (start, mid, end) ----
  for (int t = 0; t < 3; ++t) {
    for (int i = tid; i < S_ * D_; i += 256) {
      const int s = i >> 7, d = i & 127;
      long row; const float* tab;
      if (t == 0)      { row = news_ids[b];         tab = news_table; }
      else if (t == 1) { row = mid_idx[b * S_ + s]; tab = mid_table; }
      else             { row = end_idx[b * S_ + s]; tab = news_table; }
      x16[s][d] = (_Float16)tab[row * D_ + d];
      h16[s][d] = (_Float16)hbuf[s][d];
    }
    __syncthreads();

    for (int nt = wave; nt < 24; nt += 8) {
      const int nbg = nt * 16;
      v8f acc = {};
      v8f acc2 = {};
#pragma unroll
      for (int kk = 0; kk < 4; ++kk) {
        v16h ax = load_a_lds(&x16[0][0], D_, 0, kk * 32, lane);
        v16h bi = load_b_f16(wih16, D_, nbg, kk * 32, lane);
        acc = WMMA_F16(acc, ax, bi);
        v16h ah = load_a_lds(&h16[0][0], D_, 0, kk * 32, lane);
        v16h bh = load_b_f16(whh16, D_, nbg, kk * 32, lane);
        acc2 = WMMA_F16(acc2, ah, bh);
      }
      const float bih = gru_bih[nbg + n], bhh = gru_bhh[nbg + n];
#pragma unroll
      for (int v = 0; v < 8; ++v) {
        const int m = v + 8 * (lane >> 4);
        const float gi = acc[v] + bih;
        const float gh = acc2[v] + bhh;
        if (nbg + n < 2 * D_) G[m][nbg + n] = gi + gh;              // r,z preacts
        else { G[m][nbg + n] = gi; hn[m][nbg + n - 2 * D_] = gh; }  // inn | hn
      }
    }
    __syncthreads();
    for (int i = tid; i < S_ * D_; i += 256) {
      const int s = i >> 7, j = i & 127;
      const float r = 1.f / (1.f + expf(-G[s][j]));
      const float z = 1.f / (1.f + expf(-G[s][D_ + j]));
      const float nn = tanhf(G[s][2 * D_ + j] + r * hn[s][j]);
      hbuf[s][j] = (1.f - z) * nn + z * hbuf[s][j];
    }
    __syncthreads();
  }

  // ---- self-attention over S=16 (enc = hbuf) ----
  for (int i = tid; i < S_ * D_; i += 256) {
    const int s = i >> 7, d = i & 127;
    h16[s][d] = (_Float16)hbuf[s][d];
  }
  __syncthreads();

  for (int nt = wave; nt < 24; nt += 8) {  // QKV: 16x384 GEMM
    const int nbg = nt * 16;
    v8f acc = {};
#pragma unroll
    for (int kk = 0; kk < 4; ++kk) {
      v16h a = load_a_lds(&h16[0][0], D_, 0, kk * 32, lane);
      v16h bf = load_b_f16(inw16, D_, nbg, kk * 32, lane);
      acc = WMMA_F16(acc, a, bf);
    }
    const float bi = in_b[nbg + n];
#pragma unroll
    for (int v = 0; v < 8; ++v) G[v + 8 * (lane >> 4)][nbg + n] = acc[v] + bi;
  }
  __syncthreads();

  for (int idx = tid; idx < H_ * S_ * S_; idx += 256) {
    const int h = idx >> 8, i = (idx >> 4) & 15, j = idx & 15;
    float a = 0.f;
    for (int d = 0; d < 32; ++d) a += G[i][h * 32 + d] * G[j][D_ + h * 32 + d];
    sc[idx] = a * 0.17677669529663687f;
  }
  __syncthreads();
  if (tid < H_ * S_) {
    const int base = tid * S_;
    float mx = -1e30f;
    for (int j = 0; j < S_; ++j) mx = fmaxf(mx, sc[base + j]);
    float sum = 0.f;
    for (int j = 0; j < S_; ++j) { const float e = expf(sc[base + j] - mx); sc[base + j] = e; sum += e; }
    const float inv = 1.f / sum;
    for (int j = 0; j < S_; ++j) sc[base + j] *= inv;
  }
  __syncthreads();
  for (int i = tid; i < S_ * D_; i += 256) {
    const int s = i >> 7, d = i & 127, h = d >> 5;
    float a = 0.f;
    for (int j = 0; j < S_; ++j) a += sc[(h * S_ + s) * S_ + j] * G[j][2 * D_ + d];
    o16[s][d] = (_Float16)a;
  }
  __syncthreads();

  {  // output projection + residual
    const int nbg = wave * 16;
    v8f acc = {};
#pragma unroll
    for (int kk = 0; kk < 4; ++kk) {
      v16h a = load_a_lds(&o16[0][0], D_, 0, kk * 32, lane);
      v16h bf = load_b_f16(outw16, D_, nbg, kk * 32, lane);
      acc = WMMA_F16(acc, a, bf);
    }
    const float bo = out_b[nbg + n];
#pragma unroll
    for (int v = 0; v < 8; ++v) {
      const int m = v + 8 * (lane >> 4);
      hn[m][nbg + n] = hbuf[m][nbg + n] + acc[v] + bo;  // y = enc + attn
    }
  }
  __syncthreads();

  if (tid < S_) {
    float m_ = 0.f;
    for (int d = 0; d < D_; ++d) m_ += hn[tid][d];
    m_ *= (1.f / D_);
    float v_ = 0.f;
    for (int d = 0; d < D_; ++d) { const float t0 = hn[tid][d] - m_; v_ += t0 * t0; }
    v_ *= (1.f / D_);
    mu[tid] = m_;
    rs[tid] = rsqrtf(v_ + 1e-5f);
  }
  __syncthreads();
  if (tid < D_) {
    float a = 0.f;
    for (int s = 0; s < S_; ++s) a += (hn[s][tid] - mu[s]) * rs[s];
    eout[b * D_ + tid] = ln_g[tid] * (a * (1.f / S_)) + ln_b[tid];
  }
}

// ---------------- Kernel 4: meta-path gating + classifier ----------------------
__global__ __launch_bounds__(128) void combine_kernel(
    const float* __restrict__ semantic, const float* __restrict__ e0,
    const float* __restrict__ e1,
    const float* __restrict__ mp_w1, const float* __restrict__ mp_b1,
    const float* __restrict__ mp_w2, const float* __restrict__ mp_b2,
    const float* __restrict__ cls_w1, const float* __restrict__ cls_b1,
    const float* __restrict__ cls_w2, const float* __restrict__ cls_b2,
    float* __restrict__ out) {
  __shared__ float hid[2][64];
  __shared__ float scv[2];
  __shared__ float comb[2 * D_];
  __shared__ float hcls[D_];
  const int b = blockIdx.x, tid = threadIdx.x;
  const float* es[2] = {e0 + b * D_, e1 + b * D_};

  for (int p = 0; p < 2; ++p) {
    if (tid < 64) {
      const float* wr = mp_w1 + (long)tid * D_;
      float a = mp_b1[tid];
      for (int d = 0; d < D_; ++d) a += es[p][d] * wr[d];
      hid[p][tid] = tanhf(a);
    }
  }
  __syncthreads();
  if (tid < 2) {
    float a = mp_b2[0];
    for (int j = 0; j < 64; ++j) a += hid[tid][j] * mp_w2[j];
    scv[tid] = a;
  }
  __syncthreads();
  const float mx = fmaxf(scv[0], scv[1]);
  float w0 = expf(scv[0] - mx), w1 = expf(scv[1] - mx);
  const float inv = 1.f / (w0 + w1);
  w0 *= inv; w1 *= inv;
  if (tid < D_) {
    comb[tid] = semantic[b * D_ + tid];
    comb[D_ + tid] = w0 * es[0][tid] + w1 * es[1][tid];
  }
  __syncthreads();
  if (tid < D_) {
    const float* wr = cls_w1 + (long)tid * (2 * D_);
    float a = cls_b1[tid];
    for (int d = 0; d < 2 * D_; ++d) a += comb[d] * wr[d];
    hcls[tid] = fmaxf(a, 0.f);
  }
  __syncthreads();
  if (tid < 2) {
    const float* wr = cls_w2 + (long)tid * D_;
    float a = cls_b2[tid];
    for (int d = 0; d < D_; ++d) a += hcls[d] * wr[d];
    out[b * 2 + tid] = a;
  }
}

// ---------------- launch -------------------------------------------------------
extern "C" void kernel_launch(void* const* d_in, const int* in_sizes, int n_in,
                              void* d_out, int out_size, void* d_ws, size_t ws_size,
                              hipStream_t stream) {
  (void)in_sizes; (void)n_in; (void)out_size; (void)ws_size;
  const int*   news_ids     = (const int*)d_in[0];
  const int*   nwi          = (const int*)d_in[1];
  const int*   wctx         = (const int*)d_in[2];
  const int*   nsn_src      = (const int*)d_in[3];
  const int*   nsn_end      = (const int*)d_in[4];
  const int*   nun_usr      = (const int*)d_in[5];
  const int*   nun_end      = (const int*)d_in[6];
  const float* word_table   = (const float*)d_in[7];
  const float* news_table   = (const float*)d_in[8];
  const float* user_table   = (const float*)d_in[9];
  const float* source_table = (const float*)d_in[10];
  const float* wa_in_w  = (const float*)d_in[11];
  const float* wa_in_b  = (const float*)d_in[12];
  const float* wa_out_w = (const float*)d_in[13];
  const float* wa_out_b = (const float*)d_in[14];
  const float* conv_w3 = (const float*)d_in[15];
  const float* conv_b3 = (const float*)d_in[16];
  const float* conv_w4 = (const float*)d_in[17];
  const float* conv_b4 = (const float*)d_in[18];
  const float* conv_w5 = (const float*)d_in[19];
  const float* conv_b5 = (const float*)d_in[20];
  const float* fc_w = (const float*)d_in[21];
  const float* fc_b = (const float*)d_in[22];
  const float* nsn_w[10]; const float* nun_w[10];
  for (int i = 0; i < 10; ++i) { nsn_w[i] = (const float*)d_in[23 + i]; nun_w[i] = (const float*)d_in[33 + i]; }
  const float* mp_w1  = (const float*)d_in[43];
  const float* mp_b1  = (const float*)d_in[44];
  const float* mp_w2  = (const float*)d_in[45];
  const float* mp_b2  = (const float*)d_in[46];
  const float* cls_w1 = (const float*)d_in[47];
  const float* cls_b1 = (const float*)d_in[48];
  const float* cls_w2 = (const float*)d_in[49];
  const float* cls_b2 = (const float*)d_in[50];

  // workspace carve-up (256B aligned regions)
  char* base = (char*)d_ws;
  size_t off = 0;
  auto carve = [&](size_t bytes) -> char* {
    char* p = base + off;
    off += (bytes + 255) & ~(size_t)255;
    return p;
  };
  _Float16* doch = (_Float16*)carve(sizeof(_Float16) * B_ * L_ * D_);
  _Float16* kv16 = (_Float16*)carve(sizeof(_Float16) * 2 * D_ * D_);
  _Float16* cp3  = (_Float16*)carve(sizeof(_Float16) * C_ * D_ * 3);
  _Float16* cp4  = (_Float16*)carve(sizeof(_Float16) * C_ * D_ * 4);
  _Float16* cp5  = (_Float16*)carve(sizeof(_Float16) * C_ * D_ * 5);
  _Float16* mod0 = (_Float16*)carve(sizeof(_Float16) * 10 * D_ * D_);
  _Float16* mod1 = (_Float16*)carve(sizeof(_Float16) * 10 * D_ * D_);
  float* semantic = (float*)carve(sizeof(float) * B_ * D_);
  float* nsn_e    = (float*)carve(sizeof(float) * B_ * D_);
  float* nun_e    = (float*)carve(sizeof(float) * B_ * D_);

  const int TPB = 256;
  // weight pre-conversion (f32 -> f16, plus conv im2col repack)
  cvt_f16_kernel<<<(2 * D_ * D_ + TPB - 1) / TPB, TPB, 0, stream>>>(wa_in_w + D_ * D_, kv16, 2 * D_ * D_);
  repack_conv_kernel<<<(C_ * D_ * 3 + TPB - 1) / TPB, TPB, 0, stream>>>(conv_w3, cp3, 3);
  repack_conv_kernel<<<(C_ * D_ * 4 + TPB - 1) / TPB, TPB, 0, stream>>>(conv_w4, cp4, 4);
  repack_conv_kernel<<<(C_ * D_ * 5 + TPB - 1) / TPB, TPB, 0, stream>>>(conv_w5, cp5, 5);
  for (int m = 0; m < 2; ++m) {
    const float* const* wm = m ? nun_w : nsn_w;
    _Float16* dst = m ? mod1 : mod0;
    cvt_f16_kernel<<<(3 * D_ * D_ + TPB - 1) / TPB, TPB, 0, stream>>>(wm[0], dst + 0 * D_ * D_, 3 * D_ * D_);
    cvt_f16_kernel<<<(3 * D_ * D_ + TPB - 1) / TPB, TPB, 0, stream>>>(wm[1], dst + 3 * D_ * D_, 3 * D_ * D_);
    cvt_f16_kernel<<<(3 * D_ * D_ + TPB - 1) / TPB, TPB, 0, stream>>>(wm[4], dst + 6 * D_ * D_, 3 * D_ * D_);
    cvt_f16_kernel<<<(1 * D_ * D_ + TPB - 1) / TPB, TPB, 0, stream>>>(wm[6], dst + 9 * D_ * D_, 1 * D_ * D_);
  }

  fine_attn_kernel<<<B_ * L_, 256, 0, stream>>>(
      nwi, wctx, word_table, news_table, wa_in_w, wa_in_b, kv16, wa_out_w, wa_out_b, doch);

  conv_pool_fc_kernel<<<B_, 256, 0, stream>>>(
      doch, cp3, conv_b3, cp4, conv_b4, cp5, conv_b5, fc_w, fc_b, semantic);

  metapath_kernel<<<B_, 256, 0, stream>>>(
      news_ids, nsn_src, nsn_end, news_table, source_table, mod0,
      nsn_w[2], nsn_w[3], nsn_w[5], nsn_w[7], nsn_w[8], nsn_w[9], nsn_e);

  metapath_kernel<<<B_, 256, 0, stream>>>(
      news_ids, nun_usr, nun_end, news_table, user_table, mod1,
      nun_w[2], nun_w[3], nun_w[5], nun_w[7], nun_w[8], nun_w[9], nun_e);

  combine_kernel<<<B_, 128, 0, stream>>>(
      semantic, nsn_e, nun_e, mp_w1, mp_b1, mp_w2, mp_b2,
      cls_w1, cls_b1, cls_w2, cls_b2, (float*)d_out);
}
